// VectorQuantizer_8847632630303
// MI455X (gfx1250) — compile-verified, matrix operand order checked by reading, not run on hardware
//
#include <hip/hip_runtime.h>
#include <hip/hip_bf16.h>
#include <float.h>
#include <stdint.h>

typedef __attribute__((ext_vector_type(16))) __bf16        v16bf;
typedef __attribute__((ext_vector_type(8)))  float         v8f;
typedef __attribute__((ext_vector_type(8)))  unsigned int  v8u;

#define C_DIM    256
#define K_CODES  1024
#define M_TILE   128          // rows per block (8 waves x 16)
#define K_CHUNK  64           // codes staged in LDS per iteration
#define NUM_CHUNKS (K_CODES / K_CHUNK)
#define PITCH    264          // bf16 elems per LDS row (256 + 8 pad -> bank-conflict free)

__device__ __forceinline__ unsigned short f2bf(float f) {
    unsigned u = __float_as_uint(f);
    unsigned r = u + 0x7FFFu + ((u >> 16) & 1u);   // round-to-nearest-even
    return (unsigned short)(r >> 16);
}

// channel offset inside a 32-wide k-step for VGPR v, lane-half h
// (CDNA5 16-bit A/B fragment layout: v0..3 -> K=2v,2v+1 ; v4..7 -> K=16+2(v-4) ; +8 for upper half)
__device__ __forceinline__ int chOff(int v, int h) {
    return 8 * h + (v < 4 ? 2 * v : 16 + 2 * (v - 4));
}

// ---------------- kernel 0: codebook fp32 -> bf16, e_sq = ||e||^2 ----------------
__global__ void vq_prep(const float* __restrict__ cb,
                        unsigned short* __restrict__ cb_bf,
                        float* __restrict__ esq) {
    const int code = blockIdx.x;
    const int t = threadIdx.x;
    float v = cb[code * C_DIM + t];
    cb_bf[code * C_DIM + t] = f2bf(v);
    float p = v * v;
    for (int off = 16; off > 0; off >>= 1) p += __shfl_xor(p, off, 32);
    __shared__ float red[8];
    if ((t & 31) == 0) red[t >> 5] = p;
    __syncthreads();
    if (t == 0) {
        float s = 0.f;
#pragma unroll
        for (int i = 0; i < 8; ++i) s += red[i];
        esq[code] = s;
    }
}

// ---------------- kernel 1: fused bf16-WMMA GEMM + argmin ----------------
__global__ __launch_bounds__(256)
void vq_argmin(const float* __restrict__ x,
               const unsigned short* __restrict__ cb_bf,
               const float* __restrict__ esq,
               int* __restrict__ out_idx) {
    __shared__ unsigned short sB[K_CHUNK * PITCH];   // 33,792 bytes

    const int t    = threadIdx.x;
    const int wave = t >> 5;
    const int lane = t & 31;
    const int h    = lane >> 4;
    const int l16  = lane & 15;
    const long long rowBase = (long long)blockIdx.x * M_TILE + wave * 16;

    // ---- load this wave's 16 rows of x as resident bf16 A fragments ----
    v16bf a[8];
    {
        const float* xrow = x + (rowBase + l16) * C_DIM;
#pragma unroll
        for (int f = 0; f < 8; ++f) {
            v8u au;
#pragma unroll
            for (int v = 0; v < 8; ++v) {
                const int ch = 32 * f + chOff(v, h);
                float2 p = *(const float2*)(xrow + ch);
                au[v] = (unsigned)f2bf(p.x) | ((unsigned)f2bf(p.y) << 16);
            }
            a[f] = __builtin_bit_cast(v16bf, au);
        }
    }

    // per-lane running best over the columns this lane observes (n == l16 mod 16)
    float bestS[8];
    int   bestI[8];
#pragma unroll
    for (int v = 0; v < 8; ++v) { bestS[v] = FLT_MAX; bestI[v] = 0; }

    for (int ck = 0; ck < NUM_CHUNKS; ++ck) {
        const int codeBase = ck * K_CHUNK;

        // ---- stage 64 bf16 codebook rows into LDS via CDNA5 async global->LDS copy ----
        // (no VGPR round trip: tracked by ASYNCcnt, not LOADcnt/DScnt)
        {
            const char* src = (const char*)(cb_bf + (size_t)codeBase * C_DIM);
#pragma unroll
            for (int i = 0; i < 8; ++i) {                 // 8 x 16B per thread
                const int linear = t + 256 * i;           // uint4 index; 32 per row
                const int r = linear >> 5;
                const int c = linear & 31;
                unsigned ldsOff = (unsigned)(uintptr_t)(&sB[r * PITCH + c * 8]);
                const void* g = (const void*)(src + (size_t)linear * 16);
                asm volatile("global_load_async_to_lds_b128 %0, %1, off"
                             :: "v"(ldsOff), "v"(g)
                             : "memory");
            }
            asm volatile("s_wait_asynccnt 0" ::: "memory");
        }
        __syncthreads();

#pragma unroll
        for (int tile = 0; tile < K_CHUNK / 16; ++tile) {
            const int nLocal  = tile * 16 + l16;
            const int nGlobal = codeBase + nLocal;

            // preload the whole tile's B fragments, then run the WMMA chain back-to-back
            v16bf bfrag[8];
#pragma unroll
            for (int f = 0; f < 8; ++f) {
                v8u bu;
#pragma unroll
                for (int v = 0; v < 8; ++v) {
                    const int cch = 32 * f + chOff(v, h);
                    bu[v] = *(const unsigned*)(&sB[nLocal * PITCH + cch]);
                }
                bfrag[f] = __builtin_bit_cast(v16bf, bu);
            }

            v8f acc = {};
#pragma unroll
            for (int f = 0; f < 8; ++f) {
                acc = __builtin_amdgcn_wmma_f32_16x16x32_bf16(
                          false, a[f], false, bfrag[f], (short)0, acc, false, false);
            }

            // branchless per-lane running-min update (no cross-lane traffic here)
            const float e2 = esq[nGlobal];
#pragma unroll
            for (int v = 0; v < 8; ++v) {
                const float s = fmaf(-2.0f, acc[v], e2);  // x_sq constant per row: irrelevant to argmin
                const bool  p = s < bestS[v];
                bestS[v] = p ? s : bestS[v];
                bestI[v] = p ? nGlobal : bestI[v];
            }
        }
        __syncthreads();
    }

    // ---- single final cross-lane argmin over the 16 column-lanes of each half ----
#pragma unroll
    for (int v = 0; v < 8; ++v) {
        float s  = bestS[v];
        int   id = bestI[v];
#pragma unroll
        for (int off = 1; off < 16; off <<= 1) {
            const float so = __shfl_xor(s, off, 32);
            const int   io = __shfl_xor(id, off, 32);
            const bool  p  = (so < s) || (so == s && io < id);
            s  = p ? so : s;
            id = p ? io : id;
        }
        bestI[v] = id;
    }

    // D-layout: VGPR v holds row m = v + 8*h; lanes l16==0 of each half write 8 rows each
    if (l16 == 0) {
#pragma unroll
        for (int v = 0; v < 8; ++v) {
            out_idx[rowBase + v + 8 * h] = bestI[v];
        }
    }
}

// ---------------- kernel 2: zq gather + float-cast indices ----------------
__global__ void vq_gather(const float* __restrict__ cb,
                          const int* __restrict__ idx,
                          float* __restrict__ zq,
                          float* __restrict__ fidx,
                          int nRows) {
    const int t = threadIdx.x;
    const long long row = (long long)blockIdx.x * 4 + (t >> 6);  // 4 rows / block
    const int c4 = t & 63;                                       // 64 float4 per row
    if (row >= nRows) return;
    const int k = idx[row];
    float4 val = ((const float4*)(cb + (long long)k * C_DIM))[c4];
    ((float4*)(zq + row * C_DIM))[c4] = val;
    if (c4 == 0) fidx[row] = (float)k;
}

extern "C" void kernel_launch(void* const* d_in, const int* in_sizes, int n_in,
                              void* d_out, int out_size, void* d_ws, size_t ws_size,
                              hipStream_t stream) {
    const float* x  = (const float*)d_in[0];
    const float* cb = (const float*)d_in[1];
    const int N = in_sizes[0] / C_DIM;   // 131072
    const int K = in_sizes[1] / C_DIM;   // 1024

    // workspace layout: [bf16 codebook | e_sq | int indices]
    unsigned short* cb_bf = (unsigned short*)d_ws;
    float* esq = (float*)((char*)d_ws + (size_t)K * C_DIM * sizeof(unsigned short));
    int*   idx = (int*)((char*)esq + (size_t)K * sizeof(float));

    float* zq   = (float*)d_out;
    float* fidx = zq + (size_t)N * C_DIM;

    vq_prep  <<<K,            256, 0, stream>>>(cb, cb_bf, esq);
    vq_argmin<<<N / M_TILE,   256, 0, stream>>>(x, cb_bf, esq, idx);
    vq_gather<<<(N + 3) / 4,  256, 0, stream>>>(cb, idx, zq, fidx, N);
}
